// GAT_DSSE_BiLevel_55989193671254
// MI455X (gfx1250) — matrix operand order repaired and checked by reading, not compiled, not run on hardware
//
#include <hip/hip_runtime.h>

#define N_NODES 50000
#define NE      800000
#define DF      8
#define DH      32
#define NH      4
#define HC      128
#define DE      4
#define DDIM    256
#define DOUT    4
#define KPAD    32
#define NLAY    3

typedef __attribute__((ext_vector_type(16))) __bf16 v16bf;
typedef __attribute__((ext_vector_type(8)))  float  v8f;

// ---- convert f32 weight [krows, ncols] -> bf16 [KPAD, ncols], zero-padded ----
__global__ void k_convw(const float* __restrict__ W, __bf16* __restrict__ Wb,
                        int krows, int ncols) {
  int t = blockIdx.x * 256 + threadIdx.x;
  int total = KPAD * ncols;
  if (t >= total) return;
  int k = t / ncols;
  Wb[t] = (k < krows) ? (__bf16)W[t] : (__bf16)0.0f;
}

// ---- A-fragment loader: row-major f32 row -> 16x32 bf16 A layout (ISA 7.12.2) ----
__device__ __forceinline__ v16bf load_a_bf16(const float* __restrict__ xrow,
                                             int kdim, int hi) {
  float f[16];
  const int k0 = 8 * hi;           // halfs 0..7  <- K = k0..k0+7
  if (k0 + 8 <= kdim) {
    const float4 u = *(const float4*)(xrow + k0);
    const float4 v = *(const float4*)(xrow + k0 + 4);
    f[0] = u.x; f[1] = u.y; f[2] = u.z; f[3] = u.w;
    f[4] = v.x; f[5] = v.y; f[6] = v.z; f[7] = v.w;
  } else {
#pragma unroll
    for (int i = 0; i < 8; ++i) f[i] = 0.0f;
  }
  const int k1 = 16 + 8 * hi;      // halfs 8..15 <- K = k1..k1+7
  if (k1 + 8 <= kdim) {
    const float4 u = *(const float4*)(xrow + k1);
    const float4 v = *(const float4*)(xrow + k1 + 4);
    f[8] = u.x; f[9] = u.y; f[10] = u.z; f[11] = u.w;
    f[12] = v.x; f[13] = v.y; f[14] = v.z; f[15] = v.w;
  } else {
#pragma unroll
    for (int i = 8; i < 16; ++i) f[i] = 0.0f;
  }
  v16bf a;
#pragma unroll
  for (int i = 0; i < 16; ++i) a[i] = (__bf16)f[i];
  return a;
}

// ---- Y[N,HC] = X[N,kdim] @ Wb[KPAD,HC] + bias ; one wave = one 16x16 tile ----
__global__ void k_gemm_hc(const float* __restrict__ X, int kdim,
                          const __bf16* __restrict__ Wb,
                          const float* __restrict__ bias,
                          float* __restrict__ Y) {
  const int wave = threadIdx.x >> 5;       // 8 waves: coltiles 0..7 (HC=128)
  const int lane = threadIdx.x & 31;
  const int m    = lane & 15;
  const int hi   = lane >> 4;
  const int row0 = blockIdx.x * 16;
  v16bf a = load_a_bf16(X + (size_t)(row0 + m) * kdim, kdim, hi);
  const int n = wave * 16 + m;
  v16bf b;
#pragma unroll
  for (int i = 0; i < 16; ++i) b[i] = Wb[((i & 7) + 8 * hi + 16 * (i >> 3)) * HC + n];
  v8f c = {};
  c = __builtin_amdgcn_wmma_f32_16x16x32_bf16(false, a, false, b, (short)0, c, false, false);
  const float bv = bias[n];
#pragma unroll
  for (int v = 0; v < 8; ++v)
    Y[(size_t)(row0 + v + 8 * hi) * HC + n] = c[v] + bv;
}

// ---- per-node per-head L2 norms of xl rows ----
__global__ void k_norms(const float* __restrict__ xl, float* __restrict__ nrm) {
  const int wave = threadIdx.x >> 5, lane = threadIdx.x & 31;
  const int n = blockIdx.x * 8 + wave;
  const float4 v4 = *(const float4*)(xl + (size_t)n * HC + lane * 4);
  float p = v4.x * v4.x + v4.y * v4.y + v4.z * v4.z + v4.w * v4.w;
  p += __shfl_xor(p, 1, 8);
  p += __shfl_xor(p, 2, 8);
  p += __shfl_xor(p, 4, 8);
  if ((lane & 7) == 0) nrm[(size_t)n * NH + (lane >> 3)] = sqrtf(p);
}

// ================= CSR build (once; edge list is layer-invariant) =============
__global__ void k_count(const long long* __restrict__ ei, int* __restrict__ deg) {
  const int e = blockIdx.x * 256 + threadIdx.x;      // NE/256 exact
  atomicAdd(&deg[(int)ei[NE + e]], 1);
}

__global__ void k_scan1(const int* __restrict__ deg, int* __restrict__ rowptr,
                        int* __restrict__ bsum) {
  __shared__ int sm[256];
  const int t = threadIdx.x, g = blockIdx.x * 256 + t;
  const int v = (g < N_NODES) ? deg[g] : 0;
  sm[t] = v;
  __syncthreads();
  for (int o = 1; o < 256; o <<= 1) {
    int x = (t >= o) ? sm[t - o] : 0;
    __syncthreads();
    sm[t] += x;
    __syncthreads();
  }
  if (g < N_NODES) rowptr[g] = sm[t] - v;            // exclusive within block
  if (t == 255) bsum[blockIdx.x] = sm[255];
}

__global__ void k_scan2(int* __restrict__ bsum, int nb) {   // single block
  __shared__ int sm[256];
  const int t = threadIdx.x;
  const int v = (t < nb) ? bsum[t] : 0;
  sm[t] = v;
  __syncthreads();
  for (int o = 1; o < 256; o <<= 1) {
    int x = (t >= o) ? sm[t - o] : 0;
    __syncthreads();
    sm[t] += x;
    __syncthreads();
  }
  if (t < nb) bsum[t] = sm[t] - v;                   // exclusive block offsets
}

__global__ void k_scan3(int* __restrict__ rowptr, const int* __restrict__ bsum) {
  const int g = blockIdx.x * 256 + threadIdx.x;
  if (g < N_NODES) rowptr[g] += bsum[blockIdx.x];
  if (g == 0) rowptr[N_NODES] = NE;
}

__global__ void k_bucket(const long long* __restrict__ ei,
                         const int* __restrict__ rowptr, int* __restrict__ cur,
                         int* __restrict__ cidx, int* __restrict__ csrc) {
  const int e = blockIdx.x * 256 + threadIdx.x;
  const int d = (int)ei[NE + e];
  const int p = rowptr[d] + atomicAdd(&cur[d], 1);
  cidx[p] = e;
  csrc[p] = (int)ei[e];
}

// canonicalize per-node lists (sort by edge id) -> deterministic FP order
__global__ void k_sortlists(const int* __restrict__ rowptr,
                            int* __restrict__ cidx, int* __restrict__ csrc) {
  const int n = blockIdx.x * 256 + threadIdx.x;
  if (n >= N_NODES) return;
  const int st = rowptr[n], en = rowptr[n + 1];
  for (int i = st + 1; i < en; ++i) {
    const int ke = cidx[i], ks = csrc[i];
    int j = i - 1;
    while (j >= st && cidx[j] > ke) {
      cidx[j + 1] = cidx[j];
      csrc[j + 1] = csrc[j];
      --j;
    }
    cidx[j + 1] = ke;
    csrc[j + 1] = ks;
  }
}

// ====== fused GAT layer: wave per dst node, online softmax, no atomics ========
__global__ void k_gat_node(const int* __restrict__ rowptr,
                           const int* __restrict__ cidx,
                           const int* __restrict__ csrc,
                           const float* __restrict__ ea,
                           const float* __restrict__ We,   // [DE*HC]
                           const float* __restrict__ att,  // [HC]
                           const float* __restrict__ xl,
                           const float* __restrict__ xr,
                           const float* __restrict__ nrm,  // [N,NH]
                           const float* __restrict__ bo,
                           float* __restrict__ hout) {
  __shared__ float sWe[DE * HC];
  __shared__ float sAtt[HC];
  const int t = threadIdx.x;
  sWe[t] = We[t];
  sWe[t + 256] = We[t + 256];
  if (t < HC) sAtt[t] = att[t];
  __syncthreads();
  const int wave = t >> 5, lane = t & 31;
  const int n = blockIdx.x * 8 + wave;           // N_NODES/8 blocks exact
  const int start = rowptr[n];
  const int deg = rowptr[n + 1] - start;
  const int hh = lane >> 3;                      // head owned for channel work
  const int c0 = lane * 4;                       // global channels c0..c0+3
  // x_i (dst transform) channels, resident for the whole node
  const float4 xi4 = *(const float4*)(xr + (size_t)n * HC + c0);
  const float xi[4] = {xi4.x, xi4.y, xi4.z, xi4.w};
  // ni per own head (reduce within 8-lane head group)
  float nip = xi[0] * xi[0] + xi[1] * xi[1] + xi[2] * xi[2] + xi[3] * xi[3];
  nip += __shfl_xor(nip, 1, 8);
  nip += __shfl_xor(nip, 2, 8);
  nip += __shfl_xor(nip, 4, 8);
  const float ni = sqrtf(nip);
  // sweep 1: mx_h = max ||x_j|| over incident edges (16B/edge)
  float mxp = 0.0f;                              // empty group -> 0 (ref semantics)
  for (int j = lane >> 2; j < deg; j += 8) {
    const int sn = csrc[start + j];
    mxp = fmaxf(mxp, nrm[(size_t)sn * NH + (lane & 3)]);
  }
  mxp = fmaxf(mxp, __shfl_xor(mxp, 4));
  mxp = fmaxf(mxp, __shfl_xor(mxp, 8));
  mxp = fmaxf(mxp, __shfl_xor(mxp, 16));         // lane l holds max for head l&3
  const float mx = __shfl(mxp, hh);              // broadcast own head's max
  const float denom = 4.0f * (ni + mx) + 1e-12f;
  // sweep 2: online softmax + weighted accumulation, single x_j gather per edge
  float M = -3.4e38f, S = 0.0f;
  float acc[4] = {0.f, 0.f, 0.f, 0.f};
  for (int j = 0; j < deg; ++j) {
    const int e = cidx[start + j];
    const int sn = csrc[start + j];
    if (j + 1 < deg)                             // cover gather latency
      __builtin_prefetch(xl + (size_t)csrc[start + j + 1] * HC + c0, 0, 0);
    const float4 xj4 = *(const float4*)(xl + (size_t)sn * HC + c0);
    const float xj[4] = {xj4.x, xj4.y, xj4.z, xj4.w};
    const float4 eav = *(const float4*)(ea + (size_t)e * DE);
    float dotp = 0.f;
#pragma unroll
    for (int q = 0; q < 4; ++q) {
      const int c = c0 + q;
      const float ef = eav.x * sWe[c] + eav.y * sWe[HC + c] +
                       eav.z * sWe[2 * HC + c] + eav.w * sWe[3 * HC + c];
      float mv = xi[q] + xj[q] + ef;
      mv = mv > 0.f ? mv : 0.2f * mv;            // GATv2 slope 0.2
      dotp += mv * sAtt[c];
    }
    dotp += __shfl_xor(dotp, 1, 8);
    dotp += __shfl_xor(dotp, 2, 8);
    dotp += __shfl_xor(dotp, 4, 8);              // head-h logit in all 8 lanes
    const float l = dotp / denom;
    const float newM = fmaxf(M, l);
    const float sc = __expf(M - newM);
    const float p = __expf(l - newM);
    S = S * sc + p;
#pragma unroll
    for (int q = 0; q < 4; ++q) acc[q] = acc[q] * sc + p * xj[q];
    M = newM;
  }
  const float invs = 1.0f / (S + 1e-16f);
#pragma unroll
  for (int q = 0; q < 4; ++q) acc[q] *= invs;
  // mean over heads: channel h*32+cc lives in lane cc/4 + 8h -> xor 8,16
#pragma unroll
  for (int q = 0; q < 4; ++q) {
    acc[q] += __shfl_xor(acc[q], 8);
    acc[q] += __shfl_xor(acc[q], 16);
  }
  if (lane < 8) {
    float* hp = hout + (size_t)n * DH + c0;      // c0 < 32 here
#pragma unroll
    for (int q = 0; q < 4; ++q) {
      float v = 0.25f * acc[q] + bo[c0 + q];
      hp[q] = v > 0.f ? v : 0.01f * v;           // outer LeakyReLU
    }
  }
}

// ---- fused MLP head: [16,32]@[32,256] (WMMA) -> LeakyReLU -> @[256,4] ----
__global__ void k_mlp(const float* __restrict__ Hin,
                      const __bf16* __restrict__ W1b,   // [KPAD, DDIM]
                      const float* __restrict__ b1,
                      const float* __restrict__ W2,     // [DDIM, DOUT]
                      const float* __restrict__ b2,
                      float* __restrict__ Out) {
  __shared__ float hid[16][DDIM];
  const int wave = threadIdx.x >> 5, lane = threadIdx.x & 31;
  const int m = lane & 15, hi = lane >> 4;
  const int row0 = blockIdx.x * 16;
  v16bf a = load_a_bf16(Hin + (size_t)(row0 + m) * DH, DH, hi);
#pragma unroll
  for (int tt = 0; tt < 2; ++tt) {
    const int n = (wave + tt * 8) * 16 + m;
    v16bf b;
#pragma unroll
    for (int i = 0; i < 16; ++i) b[i] = W1b[((i & 7) + 8 * hi + 16 * (i >> 3)) * DDIM + n];
    v8f c = {};
    c = __builtin_amdgcn_wmma_f32_16x16x32_bf16(false, a, false, b, (short)0, c, false, false);
#pragma unroll
    for (int v = 0; v < 8; ++v) {
      const float val = c[v] + b1[n];
      hid[v + 8 * hi][n] = val > 0.f ? val : 0.01f * val;
    }
  }
  __syncthreads();
  // split-K second GEMM: 256 threads = 16 rows x 4 outs x 4 K-quarters
  const int t = threadIdx.x;
  const int row = t >> 4, o = (t >> 2) & 3, q = t & 3;
  float accm = 0.f;
#pragma unroll 8
  for (int k = q * 64; k < q * 64 + 64; ++k) accm += hid[row][k] * W2[k * DOUT + o];
  accm += __shfl_xor(accm, 1, 4);
  accm += __shfl_xor(accm, 2, 4);
  if (q == 0) Out[(size_t)(row0 + row) * DOUT + o] = accm + b2[o];
}

extern "C" void kernel_launch(void* const* d_in, const int* in_sizes, int n_in,
                              void* d_out, int out_size, void* d_ws, size_t ws_size,
                              hipStream_t stream) {
  (void)in_sizes; (void)n_in; (void)out_size; (void)ws_size;
  const float*     x   = (const float*)d_in[0];
  const long long* ei  = (const long long*)d_in[1];
  const float*     ea  = (const float*)d_in[2];
  const float* Wl0 = (const float*)d_in[3];
  const float* bl0 = (const float*)d_in[4];
  const float* Wr0 = (const float*)d_in[5];
  const float* br0 = (const float*)d_in[6];
  const float* We0 = (const float*)d_in[7];
  const float* att0= (const float*)d_in[8];
  const float* bo0 = (const float*)d_in[9];
  const float* Wls = (const float*)d_in[10];
  const float* bls = (const float*)d_in[11];
  const float* Wrs = (const float*)d_in[12];
  const float* brs = (const float*)d_in[13];
  const float* Wes = (const float*)d_in[14];
  const float* atts= (const float*)d_in[15];
  const float* bos = (const float*)d_in[16];
  const float* Wd1 = (const float*)d_in[17];
  const float* bd1 = (const float*)d_in[18];
  const float* Wd2 = (const float*)d_in[19];
  const float* bd2 = (const float*)d_in[20];

  // ---- workspace carve-up (4-byte units) ----
  float* base = (float*)d_ws;
  size_t off = 0;
  float* xl   = base + off; off += (size_t)N_NODES * HC;
  float* xr   = base + off; off += (size_t)N_NODES * HC;
  float* hA   = base + off; off += (size_t)N_NODES * DH;
  float* nrm  = base + off; off += (size_t)N_NODES * NH;
  int* rowptr = (int*)(base + off); off += N_NODES + 1;
  int* cidx   = (int*)(base + off); off += NE;
  int* csrc   = (int*)(base + off); off += NE;
  int* cur    = (int*)(base + off); off += N_NODES;   // also used as deg
  int* bsum   = (int*)(base + off); off += 256;
  __bf16* wbl  = (__bf16*)(base + off);
  __bf16* wbr  = wbl + KPAD * HC;
  __bf16* wbd1 = wbr + KPAD * HC;

  const int nb = (N_NODES + 255) / 256;          // 196
  const int gemm_blocks = N_NODES / 16;          // 3125 exact
  const int node_blocks = N_NODES / 8;           // 6250 exact
  const int edge_blocks = NE / 256;              // 3125 exact

  // ---- CSR build (deterministic after sort), reused by all layers ----
  hipMemsetAsync(cur, 0, (size_t)N_NODES * 4, stream);
  k_count<<<edge_blocks, 256, 0, stream>>>(ei, cur);
  k_scan1<<<nb, 256, 0, stream>>>(cur, rowptr, bsum);
  k_scan2<<<1, 256, 0, stream>>>(bsum, nb);
  k_scan3<<<nb, 256, 0, stream>>>(rowptr, bsum);
  hipMemsetAsync(cur, 0, (size_t)N_NODES * 4, stream);
  k_bucket<<<edge_blocks, 256, 0, stream>>>(ei, rowptr, cur, cidx, csrc);
  k_sortlists<<<nb, 256, 0, stream>>>(rowptr, cidx, csrc);

  for (int L = 0; L < NLAY; ++L) {
    const float* Xin = (L == 0) ? x : hA;
    const int    K   = (L == 0) ? DF : DH;
    const float* pWl = (L == 0) ? Wl0 : Wls + (size_t)(L - 1) * DH * HC;
    const float* pbl = (L == 0) ? bl0 : bls + (size_t)(L - 1) * HC;
    const float* pWr = (L == 0) ? Wr0 : Wrs + (size_t)(L - 1) * DH * HC;
    const float* pbr = (L == 0) ? br0 : brs + (size_t)(L - 1) * HC;
    const float* pWe = (L == 0) ? We0 : Wes + (size_t)(L - 1) * DE * HC;
    const float* patt= (L == 0) ? att0: atts+ (size_t)(L - 1) * NH * DH;
    const float* pbo = (L == 0) ? bo0 : bos + (size_t)(L - 1) * DH;

    k_convw<<<(KPAD * HC + 255) / 256, 256, 0, stream>>>(pWl, wbl, K, HC);
    k_convw<<<(KPAD * HC + 255) / 256, 256, 0, stream>>>(pWr, wbr, K, HC);
    k_gemm_hc<<<gemm_blocks, 256, 0, stream>>>(Xin, K, wbl, pbl, xl);
    k_gemm_hc<<<gemm_blocks, 256, 0, stream>>>(Xin, K, wbr, pbr, xr);
    k_norms<<<node_blocks, 256, 0, stream>>>(xl, nrm);
    k_gat_node<<<node_blocks, 256, 0, stream>>>(rowptr, cidx, csrc, ea, pWe, patt,
                                                xl, xr, nrm, pbo, hA);
  }

  k_convw<<<(KPAD * DDIM + 255) / 256, 256, 0, stream>>>(Wd1, wbd1, DH, DDIM);
  k_mlp<<<gemm_blocks, 256, 0, stream>>>(hA, wbd1, bd1, Wd2, bd2, (float*)d_out);
}
